// BaselineModel_67310727463165
// MI455X (gfx1250) — matrix operand order repaired and checked by reading, not compiled, not run on hardware
//
#include <hip/hip_runtime.h>
#include <math.h>

// Problem constants (from reference)
#define B_TOTAL 4096
#define SEQ_T   80
#define EMB_E   300
#define EMB_EP  320   // padded to multiple of 32 for K-tiling
#define HID_H   128

#define WAVES_PER_BLOCK 2
#define THREADS (WAVES_PER_BLOCK * 32)
#define ROWS_PER_WAVE   16
#define ROWS_PER_BLOCK  (WAVES_PER_BLOCK * ROWS_PER_WAVE)

typedef __attribute__((ext_vector_type(16))) _Float16 v16h;
typedef __attribute__((ext_vector_type(8)))  _Float16 v8h;
typedef __attribute__((ext_vector_type(8)))  float    v8f;
typedef __attribute__((ext_vector_type(4)))  float    v4f;

// B-tile (32x16 f16) fetch from a transposed [N][K] weight image in LDS.
// Layout: lane holds column N = lane&15; half j <-> K = (lane>>4)*8 + (j&7) + (j>>3)*16.
__device__ __forceinline__ v16h load_b_tile(const _Float16* wT, int Lk,
                                            int k0, int nt, int m, int hi) {
  const _Float16* row = wT + (nt * 16 + m) * Lk + k0 + (hi << 3);
  v8h b0 = *(const v8h*)(row);
  v8h b1 = *(const v8h*)(row + 16);
  v16h b;
#pragma unroll
  for (int j = 0; j < 8; ++j) { b[j] = b0[j]; b[j + 8] = b1[j]; }
  return b;
}

__global__ void __launch_bounds__(THREADS, 1)
rnn_fused_wmma_kernel(const int*   __restrict__ xg,
                      const float* __restrict__ emb,
                      const float* __restrict__ Wxh,
                      const float* __restrict__ Whh,
                      const float* __restrict__ bh,
                      const float* __restrict__ Wd,
                      const float* __restrict__ bd,
                      float* __restrict__ out) {
  extern __shared__ _Float16 smem[];
  _Float16* sWxhT = smem;                          // [H][EP]  (transposed, K-padded)
  _Float16* sWhhT = sWxhT + HID_H * EMB_EP;        // [H][H]   (transposed)
  _Float16* sH    = sWhhT + HID_H * HID_H;         // [waves][16][H] hidden staging

  const int tid = threadIdx.x;

  // Cooperative load of transposed f16 weights into LDS (K padded with zeros)
  for (int i = tid; i < HID_H * EMB_EP; i += THREADS) {
    const int n = i / EMB_EP, k = i % EMB_EP;
    sWxhT[i] = (k < EMB_E) ? (_Float16)Wxh[k * HID_H + n] : (_Float16)0.f;
  }
  for (int i = tid; i < HID_H * HID_H; i += THREADS) {
    const int n = i / HID_H, k = i % HID_H;
    sWhhT[i] = (_Float16)Whh[k * HID_H + n];
  }
  // h0 = 0
  for (int i = tid; i < WAVES_PER_BLOCK * ROWS_PER_WAVE * HID_H; i += THREADS)
    sH[i] = (_Float16)0.f;
  __syncthreads();

  const int lane  = tid & 31;
  const int wave  = tid >> 5;
  const int m     = lane & 15;   // matrix row (A) / column (B,C)
  const int hi    = lane >> 4;   // lane half selector
  const int khalf = hi * 8;      // A-layout K offset per lane half
  _Float16* sHw = sH + wave * (ROWS_PER_WAVE * HID_H);
  const int b0 = blockIdx.x * ROWS_PER_BLOCK + wave * ROWS_PER_WAVE;

  float bias[8];
#pragma unroll
  for (int nt = 0; nt < 8; ++nt) bias[nt] = bh[nt * 16 + m];

  v8f acc[8];

  for (int t = 0; t < SEQ_T; ++t) {
    const int rowIdx = xg[(b0 + m) * SEQ_T + t];
    const float* erow = emb + (size_t)rowIdx * EMB_E;
    if (t + 1 < SEQ_T) {  // prefetch next timestep's embedding row (L2-resident)
      const int nx = xg[(b0 + m) * SEQ_T + t + 1];
      __builtin_prefetch(emb + (size_t)nx * EMB_E, 0, 1);
    }

    // acc = bh (broadcast down M)
#pragma unroll
    for (int nt = 0; nt < 8; ++nt) {
      const float bv = bias[nt];
#pragma unroll
      for (int r = 0; r < 8; ++r) acc[nt][r] = bv;
    }

    // Input projection: acc += e_t(16xE) @ Wxh(ExH), K padded to 320
#pragma unroll
    for (int kt = 0; kt < EMB_EP / 32; ++kt) {
      const int k0 = kt * 32;
      v16h a;
      if (k0 + 32 <= EMB_E) {  // fully in-range K tile: 16B vector gathers
        v4f f0 = *(const v4f*)(erow + k0 + khalf);
        v4f f1 = *(const v4f*)(erow + k0 + khalf + 4);
        v4f f2 = *(const v4f*)(erow + k0 + 16 + khalf);
        v4f f3 = *(const v4f*)(erow + k0 + 16 + khalf + 4);
#pragma unroll
        for (int j = 0; j < 4; ++j) {
          a[j]      = (_Float16)f0[j];
          a[j + 4]  = (_Float16)f1[j];
          a[j + 8]  = (_Float16)f2[j];
          a[j + 12] = (_Float16)f3[j];
        }
      } else {                 // tail tile (K 288..319): per-element guard
#pragma unroll
        for (int j = 0; j < 16; ++j) {
          const int k = k0 + khalf + ((j >> 3) << 4) + (j & 7);
          a[j] = (_Float16)((k < EMB_E) ? erow[k] : 0.f);
        }
      }
#pragma unroll
      for (int nt = 0; nt < 8; ++nt) {
        v16h b = load_b_tile(sWxhT, EMB_EP, k0, nt, m, hi);
        acc[nt] = __builtin_amdgcn_wmma_f32_16x16x32_f16(
            false, a, false, b, (short)0, acc[nt], false, false);
      }
    }

    // Recurrence: acc += h_prev(16x128) @ Whh(128x128)
    asm volatile("" ::: "memory");
#pragma unroll
    for (int kt = 0; kt < HID_H / 32; ++kt) {
      const int k0 = kt * 32;
      const _Float16* hrow = sHw + m * HID_H + k0 + khalf;
      v8h h0 = *(const v8h*)(hrow);
      v8h h1 = *(const v8h*)(hrow + 16);
      v16h a;
#pragma unroll
      for (int j = 0; j < 8; ++j) { a[j] = h0[j]; a[j + 8] = h1[j]; }
#pragma unroll
      for (int nt = 0; nt < 8; ++nt) {
        v16h b = load_b_tile(sWhhT, HID_H, k0, nt, m, hi);
        acc[nt] = __builtin_amdgcn_wmma_f32_16x16x32_f16(
            false, a, false, b, (short)0, acc[nt], false, false);
      }
    }

    // h = tanh(acc); stage back to LDS in [m][k] layout for next step's A
    asm volatile("" ::: "memory");
#pragma unroll
    for (int nt = 0; nt < 8; ++nt) {
#pragma unroll
      for (int r = 0; r < 8; ++r) {
        const float hv = tanhf(acc[nt][r]);
        acc[nt][r] = hv;
        sHw[(r + (hi << 3)) * HID_H + nt * 16 + m] = (_Float16)hv;
      }
    }
    asm volatile("" ::: "memory");
  }

  // Dense(1) + sigmoid: reduce over hidden dim (N across lanes within halves)
  float part[8];
#pragma unroll
  for (int r = 0; r < 8; ++r) part[r] = 0.f;
#pragma unroll
  for (int nt = 0; nt < 8; ++nt) {
    const float wdv = Wd[nt * 16 + m];  // loaded post-loop: not live in recurrence
#pragma unroll
    for (int r = 0; r < 8; ++r) part[r] += acc[nt][r] * wdv;
  }

  const float bd0 = bd[0];
#pragma unroll
  for (int r = 0; r < 8; ++r) {
    float v = part[r];
    v += __shfl_xor(v, 1, 32);
    v += __shfl_xor(v, 2, 32);
    v += __shfl_xor(v, 4, 32);
    v += __shfl_xor(v, 8, 32);  // full sum within each 16-lane half
    if (m == 0) {               // lane 0 -> rows 0..7, lane 16 -> rows 8..15
      out[b0 + r + (hi << 3)] = 1.f / (1.f + expf(-(v + bd0)));
    }
  }
}

extern "C" void kernel_launch(void* const* d_in, const int* in_sizes, int n_in,
                              void* d_out, int out_size, void* d_ws, size_t ws_size,
                              hipStream_t stream) {
  (void)in_sizes; (void)n_in; (void)out_size; (void)d_ws; (void)ws_size;
  const int*   x   = (const int*)  d_in[0];
  const float* emb = (const float*)d_in[1];
  const float* Wxh = (const float*)d_in[2];
  const float* Whh = (const float*)d_in[3];
  const float* bh  = (const float*)d_in[4];
  const float* Wd  = (const float*)d_in[5];
  const float* bd  = (const float*)d_in[6];
  float* out = (float*)d_out;

  const size_t smem_halves = (size_t)HID_H * EMB_EP + (size_t)HID_H * HID_H +
                             (size_t)WAVES_PER_BLOCK * ROWS_PER_WAVE * HID_H;
  const size_t smem = smem_halves * sizeof(_Float16);  // 122,880 B < 320 KB/WGP

  hipFuncSetAttribute(reinterpret_cast<const void*>(rnn_fused_wmma_kernel),
                      hipFuncAttributeMaxDynamicSharedMemorySize, (int)smem);

  dim3 grid(B_TOTAL / ROWS_PER_BLOCK);  // 128 blocks x 64 threads (2 waves)
  rnn_fused_wmma_kernel<<<grid, THREADS, smem, stream>>>(x, emb, Wxh, Whh, bh,
                                                         Wd, bd, out);
}
